// Decoder_75694503624837
// MI455X (gfx1250) — compile-verified, matrix operand order checked by reading, not run on hardware
//
#include <hip/hip_runtime.h>
#include <cstdint>
#include <cstddef>

#define Bn 2048
#define Nn 30
#define Hh 256
#define Ee 256
#define WFn 32

typedef __bf16 bf16;
typedef __attribute__((ext_vector_type(16))) __bf16 v16bf;
typedef __attribute__((ext_vector_type(8)))  __bf16 v8bf;
typedef __attribute__((ext_vector_type(8)))  float  v8f;

__device__ __forceinline__ bf16 f2bf(float f) {
  unsigned u = __builtin_bit_cast(unsigned, f);
  unsigned r = u + 0x7FFFu + ((u >> 16) & 1u);   // round-to-nearest-even
  unsigned short h = (unsigned short)(r >> 16);
  return __builtin_bit_cast(bf16, h);
}
__device__ __forceinline__ float sigmoidf_(float x) { return 1.0f / (1.0f + expf(-x)); }
__device__ __forceinline__ v16bf cat16(v8bf lo, v8bf hi) {
  v16bf r;
#pragma unroll
  for (int e = 0; e < 8; ++e) { r[e] = lo[e]; r[8 + e] = hi[e]; }
  return r;
}

// ---------------------------------------------------------------------------
// WMMA GEMM, bf16 x bf16 -> f32:  C[M,N] = A[M,K] @ W[N,K]^T + bias[N]
// One wave per 32x64 output tile: 2 A-frags x 4 B-frags -> 8 WMMAs per k-step,
// B fragments staged before any WMMA so loads overlap the matrix pipe.
// 16-bit fragment layout: lane<16 -> row=lane, K=[0..7]+[16..23]
//                         lane>=16 -> row=lane-16, K=[8..15]+[24..31]
// => per lane two contiguous 16B chunks at elem offsets hi*8 and hi*8+16.
// K multiple of 32; M multiple of 32; N multiple of 64.
// ---------------------------------------------------------------------------
__global__ void wmma_gemm8(const bf16* __restrict__ A, const bf16* __restrict__ W,
                           const float* __restrict__ bias, float* __restrict__ Cf,
                           bf16* __restrict__ Cbf, int M, int N, int K) {
  const int lane = threadIdx.x;
  const int row  = lane & 15;
  const int hi   = lane >> 4;
  const int m0   = blockIdx.x * 32;
  const int n0   = blockIdx.y * 64;

  const bf16* ap0 = A + (size_t)(m0 + row) * K + hi * 8;
  const bf16* ap1 = A + (size_t)(m0 + 16 + row) * K + hi * 8;
  const bf16* wp0 = W + (size_t)(n0 + row) * K + hi * 8;

  v8f accA[4] = {{}, {}, {}, {}};
  v8f accB[4] = {{}, {}, {}, {}};
  for (int kk = 0; kk < K; kk += 32) {
    if (kk + 32 < K) {
      __builtin_prefetch(ap0 + kk + 32, 0, 1);   // global_prefetch
      __builtin_prefetch(ap1 + kk + 32, 0, 1);
    }
    // stage all fragments first: 12 b128 loads in flight before first WMMA
    v8bf a0l = *(const v8bf*)(ap0 + kk);
    v8bf a0h = *(const v8bf*)(ap0 + kk + 16);
    v8bf a1l = *(const v8bf*)(ap1 + kk);
    v8bf a1h = *(const v8bf*)(ap1 + kk + 16);
    v8bf bl[4], bh[4];
#pragma unroll
    for (int t = 0; t < 4; ++t) {
      const bf16* wp = wp0 + (size_t)t * 16 * K + kk;
      bl[t] = *(const v8bf*)(wp);
      bh[t] = *(const v8bf*)(wp + 16);
    }
    v16bf a0 = cat16(a0l, a0h);
    v16bf a1 = cat16(a1l, a1h);
#pragma unroll
    for (int t = 0; t < 4; ++t) {
      v16bf b = cat16(bl[t], bh[t]);
      accA[t] = __builtin_amdgcn_wmma_f32_16x16x32_bf16(false, a0, false, b, (short)0,
                                                        accA[t], false, false);
      accB[t] = __builtin_amdgcn_wmma_f32_16x16x32_bf16(false, a1, false, b, (short)0,
                                                        accB[t], false, false);
    }
  }
#pragma unroll
  for (int t = 0; t < 4; ++t) {
#pragma unroll
    for (int r = 0; r < 8; ++r) {
      int nn  = n0 + t * 16 + row;
      float bsv = bias[nn];
      int mmA = m0 + r + hi * 8;
      int mmB = mmA + 16;
      float vA = accA[t][r] + bsv;
      float vB = accB[t][r] + bsv;
      if (Cf) {
        Cf[(size_t)mmA * N + nn] = vA;
        Cf[(size_t)mmB * N + nn] = vB;
      }
      if (Cbf) {
        Cbf[(size_t)mmA * N + nn] = f2bf(vA);
        Cbf[(size_t)mmB * N + nn] = f2bf(vB);
      }
    }
  }
}

// ---------------------------------------------------------------------------
// prep kernels
// ---------------------------------------------------------------------------
__global__ void cvt_bf_k(const float* __restrict__ s, bf16* __restrict__ d, int n) {
  int i = blockIdx.x * blockDim.x + threadIdx.x;
  if (i < n) d[i] = f2bf(s[i]);
}
__global__ void pack_xh_w_k(const float* __restrict__ Wih, const float* __restrict__ Whh,
                            bf16* __restrict__ W) {  // [1024, 512] = [Wih | Whh]
  int i = blockIdx.x * blockDim.x + threadIdx.x;
  if (i >= 1024 * 512) return;
  int r = i >> 9, c = i & 511;
  float v = (c < 256) ? Wih[r * 256 + c] : Whh[r * 256 + (c - 256)];
  W[i] = f2bf(v);
}
__global__ void add_bias_k(const float* a, const float* b, float* o, int n) {
  int i = blockIdx.x * blockDim.x + threadIdx.x;
  if (i < n) o[i] = a[i] + b[i];
}

// ---------------------------------------------------------------------------
// per-step elementwise kernels
// ---------------------------------------------------------------------------
__global__ void build_xh_k(const float* __restrict__ dec, const float* __restrict__ h,
                           bf16* __restrict__ xh) {  // [B,512] bf16
  int i = blockIdx.x * blockDim.x + threadIdx.x;
  if (i >= Bn * 512) return;
  int b = i >> 9, c = i & 511;
  float v = (c < Hh) ? dec[(size_t)b * Ee + c] : h[(size_t)b * Hh + (c - Hh)];
  xh[i] = f2bf(v);
}

__global__ void lstm_elem_k(const float* __restrict__ gates, float* __restrict__ h,
                            float* __restrict__ c, const float* __restrict__ cou,
                            bf16* __restrict__ hc) {  // hc [B,288] bf16
  int i = blockIdx.x * blockDim.x + threadIdx.x;
  if (i >= Bn * Hh) return;
  int b = i / Hh, j = i % Hh;
  const float* g = gates + (size_t)b * 4 * Hh;
  float ig = sigmoidf_(g[j]);
  float fg = sigmoidf_(g[Hh + j]);
  float gg = tanhf(g[2 * Hh + j]);
  float og = sigmoidf_(g[3 * Hh + j]);
  float cn = fg * c[i] + ig * gg;
  float hn = og * tanhf(cn);
  c[i] = cn;
  h[i] = hn;
  hc[(size_t)b * 288 + j] = f2bf(hn);
  if (j < WFn) hc[(size_t)b * 288 + 256 + j] = f2bf(cou[(size_t)b * WFn + j]);
}

// mask / kNN bookkeeping: one wave per batch row, lane = node index
__global__ void mask_knn_k(const float* __restrict__ Dm, const int* __restrict__ idxs,
                           unsigned char* __restrict__ mask, unsigned char* __restrict__ knn,
                           unsigned char* __restrict__ full, const int* __restrict__ kminp) {
  int b = blockIdx.x, lane = threadIdx.x;
  int idx  = idxs[b];
  int kmin = *kminp;
  bool inr = lane < Nn;
  bool mk  = inr ? (mask[(size_t)b * Nn + lane] != 0) : true;
  float eb = inr ? Dm[((size_t)b * Nn + idx) * Nn + lane] : 0.0f;
  float w  = inr ? (mk ? 0.0f : eb) : -1.0f;        // weighted = Eb * ~mask
  unsigned long long notm = __ballot(inr && !mk);
  bool valid = __popcll(notm) > kmin;
  float bv = w; int bi = lane;
#pragma unroll
  for (int off = 16; off > 0; off >>= 1) {
    float ov = __shfl_xor(bv, off);
    int   oi = __shfl_xor(bi, off);
    if (ov > bv || (ov == bv && oi < bi)) { bv = ov; bi = oi; }
  }
  bool nxt = inr && (lane == bi) && valid;
  bool lm  = mk || (lane == idx);
  unsigned long long ab = __ballot(!inr || lm);
  if (((ab & 0x3FFFFFFFull) == 0x3FFFFFFFull) && lane == Nn - 1) lm = false;  // _mask_modify
  bool ko = inr ? (knn[(size_t)b * Nn + lane] != 0) : false;
  bool fm = lm || ko;
  if (inr) {
    mask[(size_t)b * Nn + lane] = lm ? 1 : 0;
    knn [(size_t)b * Nn + lane] = nxt ? 1 : 0;
    full[(size_t)b * Nn + lane] = fm ? 1 : 0;
  }
}

// glimpse attention: one wave per row; writes bf16 query for the next GEMM
__global__ void glimpse_attn_k(const float* __restrict__ q, const float* __restrict__ Eg,
                               const float* __restrict__ vg, const unsigned char* __restrict__ full,
                               bf16* __restrict__ out) {
  int b = blockIdx.x, lane = threadIdx.x;
  float qv[8], vv[8];
#pragma unroll
  for (int j = 0; j < 8; ++j) {
    int hIdx = j * 32 + lane;
    qv[j] = q[(size_t)b * Hh + hIdx];
    vv[j] = vg[hIdx];
  }
  float u[Nn];
  for (int n = 0; n < Nn; ++n) {
    const float* e = Eg + ((size_t)n * Bn + b) * Hh;
    float s = 0.0f;
#pragma unroll
    for (int j = 0; j < 8; ++j) s += vv[j] * tanhf(qv[j] + e[j * 32 + lane]);
#pragma unroll
    for (int off = 16; off > 0; off >>= 1) s += __shfl_xor(s, off);
    u[n] = s;
  }
  bool m[Nn];
  float mx = -1e30f;
  for (int n = 0; n < Nn; ++n) {
    m[n] = full[(size_t)b * Nn + n] != 0;
    if (!m[n]) mx = fmaxf(mx, u[n]);
  }
  float se = 0.0f;
  float wn[Nn];
  for (int n = 0; n < Nn; ++n) { wn[n] = m[n] ? 0.0f : expf(u[n] - mx); se += wn[n]; }
  float inv = 1.0f / se;
#pragma unroll
  for (int j = 0; j < 8; ++j) {
    float s = 0.0f;
    for (int n = 0; n < Nn; ++n)
      s += Eg[((size_t)n * Bn + b) * Hh + j * 32 + lane] * (wn[n] * inv);
    out[(size_t)b * Hh + j * 32 + lane] = f2bf(s);
  }
}

// pointer attention: masked log-softmax of 10*tanh(u), greedy argmax, write outputs
__global__ void pointer_attn_k(const float* __restrict__ q, const float* __restrict__ Ep,
                               const float* __restrict__ vp, const unsigned char* __restrict__ full,
                               float* __restrict__ out_logp, float* __restrict__ out_sel,
                               int* __restrict__ idxs, int t) {
  int b = blockIdx.x, lane = threadIdx.x;
  float qv[8], vv[8];
#pragma unroll
  for (int j = 0; j < 8; ++j) {
    int hIdx = j * 32 + lane;
    qv[j] = q[(size_t)b * Hh + hIdx];
    vv[j] = vp[hIdx];
  }
  float lg[Nn];
  for (int n = 0; n < Nn; ++n) {
    const float* e = Ep + ((size_t)n * Bn + b) * Hh;
    float s = 0.0f;
#pragma unroll
    for (int j = 0; j < 8; ++j) s += vv[j] * tanhf(qv[j] + e[j * 32 + lane]);
#pragma unroll
    for (int off = 16; off > 0; off >>= 1) s += __shfl_xor(s, off);
    lg[n] = 10.0f * tanhf(s);
  }
  bool m[Nn];
  float mx = -1e30f;
  for (int n = 0; n < Nn; ++n) {
    m[n] = full[(size_t)b * Nn + n] != 0;
    if (!m[n]) mx = fmaxf(mx, lg[n]);
  }
  float se = 0.0f;
  for (int n = 0; n < Nn; ++n) if (!m[n]) se += expf(lg[n] - mx);
  float lse = mx + logf(se);
  int amax = 0; float best = -1e38f;
  for (int n = 0; n < Nn; ++n) {
    float lp = m[n] ? -__builtin_inff() : lg[n] - lse;
    if (lane == 0) out_logp[(size_t)b * Nn * Nn + (size_t)t * Nn + n] = lp;
    if (!m[n] && lg[n] > best) { best = lg[n]; amax = n; }
  }
  if (lane == 0) {
    idxs[b] = amax;
    out_sel[(size_t)b * Nn + t] = (float)amax;
  }
}

__global__ void gather_dec_k(const float* __restrict__ emb, const int* __restrict__ idxs,
                             float* __restrict__ dec) {
  int i = blockIdx.x * blockDim.x + threadIdx.x;
  if (i >= Bn * Ee) return;
  int b = i / Ee, e = i % Ee;
  dec[i] = emb[((size_t)idxs[b] * Bn + b) * Ee + e];
}

__global__ void mask_out_k(const unsigned char* __restrict__ mask, float* __restrict__ out) {
  int i = blockIdx.x * blockDim.x + threadIdx.x;
  if (i < Bn * Nn) out[i] = mask[i] ? 1.0f : 0.0f;
}

// ---------------------------------------------------------------------------
extern "C" void kernel_launch(void* const* d_in, const int* in_sizes, int n_in,
                              void* d_out, int out_size, void* d_ws, size_t ws_size,
                              hipStream_t stream) {
  (void)in_sizes; (void)n_in; (void)out_size; (void)ws_size;
  const float* decoder_input = (const float*)d_in[0];
  const float* embedded      = (const float*)d_in[1];
  const float* h0            = (const float*)d_in[2];
  const float* c0            = (const float*)d_in[3];
  const float* context       = (const float*)d_in[4];
  const float* embed_cou     = (const float*)d_in[5];
  const float* Dm            = (const float*)d_in[6];
  const float* W_ih          = (const float*)d_in[7];
  const float* W_hh          = (const float*)d_in[8];
  const float* b_ih          = (const float*)d_in[9];
  const float* b_hh          = (const float*)d_in[10];
  const float* W_merge       = (const float*)d_in[11];
  const float* b_merge       = (const float*)d_in[12];
  const float* Wq_p          = (const float*)d_in[13];
  const float* bq_p          = (const float*)d_in[14];
  const float* Wr_p          = (const float*)d_in[15];
  const float* br_p          = (const float*)d_in[16];
  const float* v_p           = (const float*)d_in[17];
  const float* Wq_g          = (const float*)d_in[18];
  const float* bq_g          = (const float*)d_in[19];
  const float* Wr_g          = (const float*)d_in[20];
  const float* br_g          = (const float*)d_in[21];
  const float* v_g           = (const float*)d_in[22];
  const unsigned char* vmask = (const unsigned char*)d_in[23];
  const int* start_idx       = (const int*)d_in[24];
  const int* kminp           = (const int*)d_in[25];

  char* p = (char*)d_ws;
  auto alloc = [&](size_t bytes) -> char* {
    char* r = p; p += (bytes + 255) & ~(size_t)255; return r;
  };
  bf16* Wxh   = (bf16*)alloc((size_t)1024 * 512 * sizeof(bf16));
  bf16* Wm    = (bf16*)alloc((size_t)256 * 288 * sizeof(bf16));
  bf16* Wqg   = (bf16*)alloc((size_t)256 * 256 * sizeof(bf16));
  bf16* Wqp   = (bf16*)alloc((size_t)256 * 256 * sizeof(bf16));
  bf16* Wrg   = (bf16*)alloc((size_t)256 * 256 * sizeof(bf16));
  bf16* Wrp   = (bf16*)alloc((size_t)256 * 256 * sizeof(bf16));
  bf16* ctxbf = (bf16*)alloc((size_t)Nn * Bn * Hh * sizeof(bf16));
  float* bxh  = (float*)alloc(1024 * sizeof(float));
  float* Eg   = (float*)alloc((size_t)Nn * Bn * Hh * sizeof(float));
  float* Ep   = (float*)alloc((size_t)Nn * Bn * Hh * sizeof(float));
  float* hb   = (float*)alloc((size_t)Bn * Hh * sizeof(float));
  float* cb   = (float*)alloc((size_t)Bn * Hh * sizeof(float));
  float* dec  = (float*)alloc((size_t)Bn * Ee * sizeof(float));
  bf16* xh    = (bf16*)alloc((size_t)Bn * 512 * sizeof(bf16));
  float* gts  = (float*)alloc((size_t)Bn * 1024 * sizeof(float));
  bf16* hc    = (bf16*)alloc((size_t)Bn * 288 * sizeof(bf16));
  bf16* glbf  = (bf16*)alloc((size_t)Bn * Hh * sizeof(bf16));
  bf16* gl2bf = (bf16*)alloc((size_t)Bn * Hh * sizeof(bf16));
  float* qb   = (float*)alloc((size_t)Bn * Hh * sizeof(float));
  unsigned char* mask  = (unsigned char*)alloc(Bn * Nn);
  unsigned char* knn   = (unsigned char*)alloc(Bn * Nn);
  unsigned char* fullm = (unsigned char*)alloc(Bn * Nn);
  int* idxs = (int*)alloc(Bn * sizeof(int));

  float* out_logp = (float*)d_out;
  float* out_sel  = out_logp + (size_t)Bn * Nn * Nn;
  float* out_mask = out_sel + (size_t)Bn * Nn;

  const dim3 w32(32, 1, 1);

  // ---- one-time prep ----
  pack_xh_w_k<<<(1024 * 512 + 255) / 256, 256, 0, stream>>>(W_ih, W_hh, Wxh);
  add_bias_k<<<(1024 + 255) / 256, 256, 0, stream>>>(b_ih, b_hh, bxh, 1024);
  cvt_bf_k<<<(256 * 288 + 255) / 256, 256, 0, stream>>>(W_merge, Wm, 256 * 288);
  cvt_bf_k<<<(256 * 256 + 255) / 256, 256, 0, stream>>>(Wq_g, Wqg, 256 * 256);
  cvt_bf_k<<<(256 * 256 + 255) / 256, 256, 0, stream>>>(Wq_p, Wqp, 256 * 256);
  cvt_bf_k<<<(256 * 256 + 255) / 256, 256, 0, stream>>>(Wr_g, Wrg, 256 * 256);
  cvt_bf_k<<<(256 * 256 + 255) / 256, 256, 0, stream>>>(Wr_p, Wrp, 256 * 256);
  cvt_bf_k<<<(Nn * Bn * Hh + 255) / 256, 256, 0, stream>>>(context, ctxbf, Nn * Bn * Hh);

  // hoisted attention keys: E[n*B+b, o] = context[n,b,:]·Wr[o,:] + br[o]
  wmma_gemm8<<<dim3(Nn * Bn / 32, Hh / 64), w32, 0, stream>>>(
      ctxbf, Wrg, br_g, Eg, nullptr, Nn * Bn, Hh, Hh);
  wmma_gemm8<<<dim3(Nn * Bn / 32, Hh / 64), w32, 0, stream>>>(
      ctxbf, Wrp, br_p, Ep, nullptr, Nn * Bn, Hh, Hh);

  // state init
  hipMemcpyAsync(hb, h0, (size_t)Bn * Hh * sizeof(float), hipMemcpyDeviceToDevice, stream);
  hipMemcpyAsync(cb, c0, (size_t)Bn * Hh * sizeof(float), hipMemcpyDeviceToDevice, stream);
  hipMemcpyAsync(dec, decoder_input, (size_t)Bn * Ee * sizeof(float), hipMemcpyDeviceToDevice, stream);
  hipMemcpyAsync(idxs, start_idx, Bn * sizeof(int), hipMemcpyDeviceToDevice, stream);
  hipMemcpyAsync(mask, vmask, (size_t)Bn * Nn, hipMemcpyDeviceToDevice, stream);
  hipMemsetAsync(knn, 0, (size_t)Bn * Nn, stream);

  // ---- decode loop ----
  for (int t = 0; t < Nn; ++t) {
    mask_knn_k<<<Bn, w32, 0, stream>>>(Dm, idxs, mask, knn, fullm, kminp);
    build_xh_k<<<(Bn * 512 + 255) / 256, 256, 0, stream>>>(dec, hb, xh);
    wmma_gemm8<<<dim3(Bn / 32, 1024 / 64), w32, 0, stream>>>(xh, Wxh, bxh, gts, nullptr,
                                                             Bn, 1024, 512);
    lstm_elem_k<<<(Bn * Hh + 255) / 256, 256, 0, stream>>>(gts, hb, cb, embed_cou, hc);
    wmma_gemm8<<<dim3(Bn / 32, Hh / 64), w32, 0, stream>>>(hc, Wm, b_merge, nullptr, glbf,
                                                           Bn, Hh, 288);
    wmma_gemm8<<<dim3(Bn / 32, Hh / 64), w32, 0, stream>>>(glbf, Wqg, bq_g, qb, nullptr,
                                                           Bn, Hh, Hh);
    glimpse_attn_k<<<Bn, w32, 0, stream>>>(qb, Eg, v_g, fullm, gl2bf);
    wmma_gemm8<<<dim3(Bn / 32, Hh / 64), w32, 0, stream>>>(gl2bf, Wqp, bq_p, qb, nullptr,
                                                           Bn, Hh, Hh);
    pointer_attn_k<<<Bn, w32, 0, stream>>>(qb, Ep, v_p, fullm, out_logp, out_sel, idxs, t);
    gather_dec_k<<<(Bn * Ee + 255) / 256, 256, 0, stream>>>(embedded, idxs, dec);
  }
  mask_out_k<<<(Bn * Nn + 255) / 256, 256, 0, stream>>>(mask, out_mask);
}
